// BmmEnsemble_26886495273137
// MI455X (gfx1250) — compile-verified, matrix operand order checked by reading, not run on hardware
//
#include <hip/hip_runtime.h>

typedef __bf16 bf16_t;
typedef __attribute__((ext_vector_type(16))) __bf16 v16bf;
typedef __attribute__((ext_vector_type(8)))  float  v8f;

#define S_NUM   4
#define E_NUM   8
#define N_ATOMS 50000
#define NPS     (N_ATOMS / S_NUM)              /* 12500 atoms per species   */
#define AEV_DIM 1008
#define AEV_PAD 1024
#define D1 256
#define D2 192
#define D3 160
#define CELU_ALPHA 0.1f

/* ---- fast path (bf16 weights pre-transposed into d_ws) ---- */
#define TILE_M  32
#define M_TILES ((NPS + TILE_M - 1) / TILE_M)  /* 391 */
#define W1T_ELEMS ((size_t)S_NUM * E_NUM * D1 * AEV_PAD)  /* 8,388,608 */
#define W2T_ELEMS ((size_t)S_NUM * E_NUM * D2 * D1)       /* 1,572,864 */
#define W3T_ELEMS ((size_t)S_NUM * E_NUM * D3 * D2)       /*   983,040 */
#define WS_NEEDED ((W1T_ELEMS + W2T_ELEMS + W3T_ELEMS) * sizeof(bf16_t))
#define SMEM_BYTES (TILE_M * AEV_PAD * 2 + TILE_M * D1 * 2 + TILE_M * D2 * 2 + TILE_M * 4)

/* ---- fallback path (LDS-staged weights, TILE_M = 16) ---- */
#define FB_TILE_M  16
#define FB_M_TILES ((NPS + FB_TILE_M - 1) / FB_TILE_M)    /* 782 */

__device__ __forceinline__ float celu_f(float x) {
    return fmaxf(x, 0.0f) +
           fminf(0.0f, CELU_ALPHA * (__expf(x * (1.0f / CELU_ALPHA)) - 1.0f));
}

// Build a 16x(bf16) WMMA fragment from two contiguous 16B chunks at p[0..7]
// and p[16..23] (the per-lane K striping of 16-bit A/B operands, wave32).
__device__ __forceinline__ v16bf load_frag16(const bf16_t* p) {
    v16bf r;
#pragma unroll
    for (int i = 0; i < 8; ++i) { r[i] = p[i]; r[i + 8] = p[16 + i]; }
    return r;
}

/* =====================  weight pre-transpose  =====================
 * Wt[mat][col][k] = (k < din) ? W[mat][k][col] : 0   (f32 -> bf16)
 */
__global__ void convert_weights_kernel(const float* __restrict__ W,
                                       bf16_t* __restrict__ Wt,
                                       int din, int dout, int kpad, int nmat)
{
    size_t total = (size_t)nmat * dout * kpad;
    for (size_t i = (size_t)blockIdx.x * blockDim.x + threadIdx.x; i < total;
         i += (size_t)gridDim.x * blockDim.x) {
        size_t mat = i / ((size_t)dout * kpad);
        size_t rem = i % ((size_t)dout * kpad);
        int col = (int)(rem / kpad);
        int k   = (int)(rem % kpad);
        float v = (k < din) ? W[mat * ((size_t)din * dout) + (size_t)k * dout + col] : 0.0f;
        Wt[i] = (bf16_t)v;
    }
}

/* =====================  fast fused-MLP kernel  ===================== */

// Bias + CELU epilogue for one (2 m-tiles x 1 n-tile) column of accumulators.
// C layout: VGPR r <-> M = r + 8*(lane>=16), N = lane&15 (within a tile).
__device__ __forceinline__ void store_col(const v8f& accLo, const v8f& accHi,
                                          const float* __restrict__ bg,
                                          bf16_t* sOut, int ldo, int col, int lane)
{
    const int mrow = (lane >> 4) * 8;
    float bv = bg[col];
#pragma unroll
    for (int r = 0; r < 8; ++r) {
        float v = celu_f(accLo[r] + bv);
        sOut[(size_t)(mrow + r) * ldo + col] = (bf16_t)v;
    }
#pragma unroll
    for (int r = 0; r < 8; ++r) {
        float v = celu_f(accHi[r] + bv);
        sOut[(size_t)(16 + mrow + r) * ldo + col] = (bf16_t)v;
    }
}

// Branch-free k-loop body: NTILES (1 or 2) n-tiles, both m-tiles of the
// 32-row A block. A: LDS bf16 (row stride lda); B: global bf16 Wt[col][KPAD].
template <int NTILES, int KPAD>
__device__ __forceinline__ void layer_body(const bf16_t* sIn, int lda,
                                           const bf16_t* __restrict__ Wt,
                                           const float* __restrict__ bg,
                                           bf16_t* sOut, int ldo,
                                           int nt0, int nt1, int lane)
{
    constexpr int KT = KPAD / 32;
    const int m  = lane & 15;
    const int kb = (lane >> 4) * 8;

    v8f acc00 = {}, acc10 = {}, acc01 = {}, acc11 = {};

    const bf16_t* a0 = sIn + (size_t)m        * lda + kb;
    const bf16_t* a1 = sIn + (size_t)(16 + m) * lda + kb;
    const bf16_t* b0 = Wt + (size_t)(nt0 * 16 + m) * KPAD + kb;
    const bf16_t* b1 = Wt + (size_t)(nt1 * 16 + m) * KPAD + kb;

    for (int kt = 0; kt < KT; ++kt) {
        const int k0 = kt * 32;
        v16bf A0 = load_frag16(a0 + k0);
        v16bf A1 = load_frag16(a1 + k0);
        v16bf B0 = load_frag16(b0 + k0);
        if ((kt + 1) < KT)                       // all lanes: covers next slab
            __builtin_prefetch(b0 + k0 + 32, 0, 1);
        acc00 = __builtin_amdgcn_wmma_f32_16x16x32_bf16(false, A0, false, B0,
                                                        (short)0, acc00, false, false);
        acc10 = __builtin_amdgcn_wmma_f32_16x16x32_bf16(false, A1, false, B0,
                                                        (short)0, acc10, false, false);
        if (NTILES == 2) {
            v16bf B1 = load_frag16(b1 + k0);
            if ((kt + 1) < KT)
                __builtin_prefetch(b1 + k0 + 32, 0, 1);
            acc01 = __builtin_amdgcn_wmma_f32_16x16x32_bf16(false, A0, false, B1,
                                                            (short)0, acc01, false, false);
            acc11 = __builtin_amdgcn_wmma_f32_16x16x32_bf16(false, A1, false, B1,
                                                            (short)0, acc11, false, false);
        }
    }

    store_col(acc00, acc10, bg, sOut, ldo, nt0 * 16 + m, lane);
    if (NTILES == 2)
        store_col(acc01, acc11, bg, sOut, ldo, nt1 * 16 + m, lane);
}

// Wave w owns n-tiles {w, w+8 (if < NT)}; the 1-vs-2 decision is a scalar
// branch (wave is readfirstlane'd), hoisted out of the k-loop.
template <int NT, int KPAD>
__device__ __forceinline__ void layer_tc(const bf16_t* sIn, int lda,
                                         const bf16_t* __restrict__ Wt,
                                         const float* __restrict__ bg,
                                         bf16_t* sOut, int ldo,
                                         int wave, int lane)
{
    if (NT >= 16) {
        layer_body<2, KPAD>(sIn, lda, Wt, bg, sOut, ldo, wave, wave + 8, lane);
    } else if (wave + 8 < NT) {            // scalar (SGPR) condition
        layer_body<2, KPAD>(sIn, lda, Wt, bg, sOut, ldo, wave, wave + 8, lane);
    } else {
        layer_body<1, KPAD>(sIn, lda, Wt, bg, sOut, ldo, wave, wave, lane);
    }
}

__global__ void __launch_bounds__(256, 1)
ani_fast_kernel(const float* __restrict__ aev, const int* __restrict__ idx,
                const bf16_t* __restrict__ W1t, const bf16_t* __restrict__ W2t,
                const bf16_t* __restrict__ W3t,
                const float* __restrict__ b1, const float* __restrict__ b2,
                const float* __restrict__ b3,
                const float* __restrict__ W4, const float* __restrict__ b4,
                float* __restrict__ out)
{
    extern __shared__ char smem[];
    bf16_t* sA   = (bf16_t*)smem;                                    // 64 KB
    bf16_t* sB1  = (bf16_t*)(smem + TILE_M * AEV_PAD * 2);           // 16 KB (h1/h3)
    bf16_t* sB2  = (bf16_t*)(smem + TILE_M * AEV_PAD * 2 + TILE_M * D1 * 2); // 12 KB
    float*  sAcc = (float*)(smem + TILE_M * AEV_PAD * 2 + TILE_M * D1 * 2 + TILE_M * D2 * 2);

    const int tid  = threadIdx.x;
    const int wave = __builtin_amdgcn_readfirstlane(tid >> 5);  // uniform SGPR
    const int lane = tid & 31;
    const int sb   = blockIdx.x / M_TILES;
    const int tm   = blockIdx.x % M_TILES;
    const int r0   = tm * TILE_M;
    const int* idxs = idx + sb * NPS;

    // Gather TILE_M AEV rows -> bf16 LDS, zero-padded K 1008 -> 1024.
    for (int i = tid; i < TILE_M * (AEV_PAD / 2); i += 256) {
        int mrow = i / (AEV_PAD / 2);
        int c    = (i % (AEV_PAD / 2)) * 2;
        float f0 = 0.0f, f1 = 0.0f;
        int row = r0 + mrow;
        if (row < NPS) {
            const float* ap = aev + (size_t)idxs[row] * AEV_DIM;
            if (c     < AEV_DIM) f0 = ap[c];
            if (c + 1 < AEV_DIM) f1 = ap[c + 1];
        }
        union { bf16_t h[2]; unsigned u; } pk;
        pk.h[0] = (bf16_t)f0; pk.h[1] = (bf16_t)f1;
        ((unsigned*)sA)[i] = pk.u;
    }
    if (tid < TILE_M) sAcc[tid] = 0.0f;
    __syncthreads();

    for (int e = 0; e < E_NUM; ++e) {
        const size_t se = (size_t)(sb * E_NUM + e);
        layer_tc<16, AEV_PAD>(sA, AEV_PAD, W1t + se * ((size_t)D1 * AEV_PAD),
                              b1 + se * D1, sB1, D1, wave, lane);
        __syncthreads();
        layer_tc<12, D1>(sB1, D1, W2t + se * ((size_t)D2 * D1),
                         b2 + se * D2, sB2, D2, wave, lane);
        __syncthreads();
        layer_tc<10, D2>(sB2, D2, W3t + se * ((size_t)D3 * D2),
                         b3 + se * D3, sB1, D3, wave, lane);
        __syncthreads();
        // Layer 4: [TILE_M x 160] x [160 x 1] + b4; 8 lanes per atom + shfl tree.
        {
            const int atom = tid >> 3;           // 0..31
            const int sub  = tid & 7;            // 0..7, 20 elements each
            const float* w4 = W4 + se * D3;
            float s = 0.0f;
            const int kbeg = sub * (D3 / 8);
#pragma unroll
            for (int k = 0; k < D3 / 8; ++k)
                s += (float)sB1[atom * D3 + kbeg + k] * w4[kbeg + k];
#pragma unroll
            for (int off = 4; off >= 1; off >>= 1)
                s += __shfl_xor(s, off, 8);
            if (sub == 0) sAcc[atom] += s + b4[se];
        }
        __syncthreads();
    }

    if (tid == 0) {
        int nvalid = NPS - r0; if (nvalid > TILE_M) nvalid = TILE_M;
        float s = 0.0f;
        for (int mrow = 0; mrow < nvalid; ++mrow) s += sAcc[mrow];
        atomicAdd(out, s * (1.0f / E_NUM));
    }
}

/* =====================  fallback kernel (LDS-staged weights)  ===================== */

template <int DIN_PAD, int DOUT>
__device__ __forceinline__ void mlp_layer_stage(const bf16_t* sIn, int lda, bf16_t* sW,
                                                const float* __restrict__ Wg,
                                                const float* __restrict__ bg, int kmax,
                                                bf16_t* sOut, int ldo,
                                                int tid, int wave, int lane)
{
    constexpr int KT = DIN_PAD / 32;
    constexpr int NT = DOUT / 16;
    const int  m    = lane & 15;
    const int  kb   = (lane >> 4) * 8;
    const int  n0   = wave * 16;
    const int  n1   = (wave + 8) * 16;
    const bool has1 = (wave + 8) < NT;     // scalar: wave is readfirstlane'd

    v8f acc0 = {}; v8f acc1 = {};

    for (int kt = 0; kt < KT; ++kt) {
        const int k0 = kt * 32;
        __syncthreads();
        for (int i = tid; i < DOUT * 16; i += 256) {
            int col = i % DOUT;
            int kp  = i / DOUT;
            int k   = k0 + 2 * kp;
            float f0 = (k     < kmax) ? Wg[(size_t)k * DOUT + col]       : 0.0f;
            float f1 = (k + 1 < kmax) ? Wg[(size_t)(k + 1) * DOUT + col] : 0.0f;
            union { bf16_t h[2]; unsigned u; } pk;
            pk.h[0] = (bf16_t)f0; pk.h[1] = (bf16_t)f1;
            ((unsigned*)sW)[col * 16 + kp] = pk.u;
        }
        if ((kt + 1) < KT)
            __builtin_prefetch(Wg + (size_t)(k0 + 32) * DOUT + tid, 0, 1);
        __syncthreads();

        v16bf a  = load_frag16(sIn + m * lda + k0 + kb);
        v16bf b0 = load_frag16(sW + (n0 + m) * 32 + kb);
        acc0 = __builtin_amdgcn_wmma_f32_16x16x32_bf16(false, a, false, b0,
                                                       (short)0, acc0, false, false);
        if (has1) {
            v16bf b1f = load_frag16(sW + (n1 + m) * 32 + kb);
            acc1 = __builtin_amdgcn_wmma_f32_16x16x32_bf16(false, a, false, b1f,
                                                           (short)0, acc1, false, false);
        }
    }

    const int mrow = (lane >> 4) * 8;
    {
        float bv = bg[n0 + m];
#pragma unroll
        for (int r = 0; r < 8; ++r) {
            float v = celu_f(acc0[r] + bv);
            sOut[(mrow + r) * ldo + (n0 + m)] = (bf16_t)v;
        }
    }
    if (has1) {
        float bv = bg[n1 + m];
#pragma unroll
        for (int r = 0; r < 8; ++r) {
            float v = celu_f(acc1[r] + bv);
            sOut[(mrow + r) * ldo + (n1 + m)] = (bf16_t)v;
        }
    }
}

__global__ void __launch_bounds__(256, 1)
ani_fallback_kernel(const float* __restrict__ aev, const int* __restrict__ idx,
                    const float* __restrict__ W1, const float* __restrict__ b1,
                    const float* __restrict__ W2, const float* __restrict__ b2,
                    const float* __restrict__ W3, const float* __restrict__ b3,
                    const float* __restrict__ W4, const float* __restrict__ b4,
                    float* __restrict__ out)
{
    __shared__ __align__(16) bf16_t sA [FB_TILE_M * AEV_PAD];
    __shared__ __align__(16) bf16_t sW [D1 * 32];
    __shared__ __align__(16) bf16_t sB1[FB_TILE_M * D1];
    __shared__ __align__(16) bf16_t sB2[FB_TILE_M * D2];
    __shared__ float sAcc[FB_TILE_M];

    const int tid  = threadIdx.x;
    const int wave = __builtin_amdgcn_readfirstlane(tid >> 5);
    const int lane = tid & 31;
    const int sb   = blockIdx.x / FB_M_TILES;
    const int tm   = blockIdx.x % FB_M_TILES;
    const int r0   = tm * FB_TILE_M;
    const int* idxs = idx + sb * NPS;

    for (int i = tid; i < FB_TILE_M * (AEV_PAD / 2); i += 256) {
        int mrow = i / (AEV_PAD / 2);
        int c    = (i % (AEV_PAD / 2)) * 2;
        float f0 = 0.0f, f1 = 0.0f;
        int row = r0 + mrow;
        if (row < NPS) {
            const float* ap = aev + (size_t)idxs[row] * AEV_DIM;
            if (c     < AEV_DIM) f0 = ap[c];
            if (c + 1 < AEV_DIM) f1 = ap[c + 1];
        }
        union { bf16_t h[2]; unsigned u; } pk;
        pk.h[0] = (bf16_t)f0; pk.h[1] = (bf16_t)f1;
        ((unsigned*)sA)[i] = pk.u;
    }
    if (tid < FB_TILE_M) sAcc[tid] = 0.0f;

    for (int e = 0; e < E_NUM; ++e) {
        const size_t se = (size_t)(sb * E_NUM + e);
        mlp_layer_stage<AEV_PAD, D1>(sA,  AEV_PAD, sW, W1 + se * (size_t)AEV_DIM * D1,
                                     b1 + se * D1, AEV_DIM, sB1, D1, tid, wave, lane);
        mlp_layer_stage<D1, D2>(sB1, D1, sW, W2 + se * (size_t)D1 * D2,
                                b2 + se * D2, D1, sB2, D2, tid, wave, lane);
        mlp_layer_stage<D2, D3>(sB2, D2, sW, W3 + se * (size_t)D2 * D3,
                                b3 + se * D3, D2, sB1, D3, tid, wave, lane);
        __syncthreads();
        if (tid < FB_TILE_M) {
            const float* w4 = W4 + se * D3;
            float s = 0.0f;
            for (int k = 0; k < D3; ++k) s += (float)sB1[tid * D3 + k] * w4[k];
            sAcc[tid] += s + b4[se];
        }
        __syncthreads();
    }

    if (tid == 0) {
        int nvalid = NPS - r0; if (nvalid > FB_TILE_M) nvalid = FB_TILE_M;
        float s = 0.0f;
        for (int mrow = 0; mrow < nvalid; ++mrow) s += sAcc[mrow];
        atomicAdd(out, s * (1.0f / E_NUM));
    }
}

__global__ void zero_out_kernel(float* out) { out[0] = 0.0f; }

extern "C" void kernel_launch(void* const* d_in, const int* in_sizes, int n_in,
                              void* d_out, int out_size, void* d_ws, size_t ws_size,
                              hipStream_t stream)
{
    (void)in_sizes; (void)n_in; (void)out_size;
    const float* aev = (const float*)d_in[0];
    /* d_in[1] = species (unused; idx already encodes the gather) */
    const int*   idx = (const int*)d_in[2];
    const float* W1  = (const float*)d_in[3];
    const float* b1  = (const float*)d_in[4];
    const float* W2  = (const float*)d_in[5];
    const float* b2  = (const float*)d_in[6];
    const float* W3  = (const float*)d_in[7];
    const float* b3  = (const float*)d_in[8];
    const float* W4  = (const float*)d_in[9];
    const float* b4  = (const float*)d_in[10];
    float* out = (float*)d_out;

    zero_out_kernel<<<dim3(1), dim3(1), 0, stream>>>(out);

    if (ws_size >= WS_NEEDED) {
        bf16_t* W1t = (bf16_t*)d_ws;
        bf16_t* W2t = W1t + W1T_ELEMS;
        bf16_t* W3t = W2t + W2T_ELEMS;
        convert_weights_kernel<<<dim3(2048), dim3(256), 0, stream>>>(
            W1, W1t, AEV_DIM, D1, AEV_PAD, S_NUM * E_NUM);
        convert_weights_kernel<<<dim3(1024), dim3(256), 0, stream>>>(
            W2, W2t, D1, D2, D1, S_NUM * E_NUM);
        convert_weights_kernel<<<dim3(1024), dim3(256), 0, stream>>>(
            W3, W3t, D2, D3, D2, S_NUM * E_NUM);
        ani_fast_kernel<<<dim3(S_NUM * M_TILES), dim3(256), SMEM_BYTES, stream>>>(
            aev, idx, W1t, W2t, W3t, b1, b2, b3, W4, b4, out);
    } else {
        ani_fallback_kernel<<<dim3(S_NUM * FB_M_TILES), dim3(256), 0, stream>>>(
            aev, idx, W1, b1, W2, b2, W3, b3, W4, b4, out);
    }
}
